// ReModel_73486890435249
// MI455X (gfx1250) — matrix operand order repaired and checked by reading, not compiled, not run on hardware
//
#include <hip/hip_runtime.h>

typedef __bf16 bf16;
typedef __attribute__((ext_vector_type(8)))  bf16  v8bf;
typedef __attribute__((ext_vector_type(16))) bf16  v16bf;
typedef __attribute__((ext_vector_type(8)))  float v8f;

#define DEV __device__ __forceinline__

constexpr int Bsz = 4, Lsz = 1024, Hsz = 768, NHd = 12, Ee = 24, Dd = 64, Rr = 97;
constexpr int BSb = 64, Gg = 12;
constexpr int INf   = 2 * Hsz + Dd;     // 1600
constexpr int Pp    = Ee * (Ee - 1);    // 552
constexpr int Ppad  = 640;              // 5*128: ha rows per doc, zero padded
constexpr int Nrow  = Bsz * Pp;         // 2208
constexpr int Nrowp = 2304;             // 18*128: padded row count
constexpr int K3    = 3 * INf;          // 4800

constexpr int BM = 128, BN = 64, BKL = 64;   // LDS K-stage 64 = 2 WMMA K-steps

DEV bf16 f2bf(float f) {
  union { float f; unsigned u; } x{f};
  unsigned r = x.u + 0x7fffu + ((x.u >> 16) & 1u);
  unsigned short h = (unsigned short)(r >> 16);
  return __builtin_bit_cast(bf16, h);
}
DEV float bf2f(bf16 b) {
  unsigned short h = __builtin_bit_cast(unsigned short, b);
  union { unsigned u; float f; } x{(unsigned)h << 16};
  return x.f;
}

// ---- CDNA5 async global->LDS copy (ASYNCcnt-tracked, ISA 08 §4) ----
DEV void async_copy_b128(unsigned lds_off, const bf16* gptr) {
  asm volatile("global_load_async_to_lds_b128 %0, %1, off"
               :: "v"(lds_off), "v"(gptr) : "memory");
}
DEV void async_wait0() {
  asm volatile("s_wait_asynccnt 0x0" ::: "memory");
}
DEV unsigned lds_off_of(const bf16* p) {
  return (unsigned)(unsigned long long)p;   // LDS aperture: offset in addr[31:0]
}

// ---- WMMA fragment helpers (CDNA5 16-bit A 16x32 / B 32x16 lane layouts) ----
DEV v16bf load_a_frag(const bf16* As, int row, int ks, int kh) {
  const v8bf* base = (const v8bf*)(As + row * BKL + ks * 32);
  v8bf lo = base[2 * kh];       // K 0..7  (lanes 0-15) / K 16..23 (lanes 16-31)
  v8bf hi = base[2 * kh + 1];   // K 8..15              / K 24..31
  v16bf a;
#pragma unroll
  for (int i = 0; i < 8; ++i) { a[i] = lo[i]; a[8 + i] = hi[i]; }
  return a;
}
DEV v16bf load_b_frag(const bf16* Bs, int nrow, int ks, int kh) {
  return *(const v16bf*)(Bs + nrow * BKL + ks * 32 + kh * 16);
}

// ---------------------------------------------------------------- hs -> bf16
__global__ __launch_bounds__(256) void prep_hs_kernel(const float* __restrict__ hs,
                                                      bf16* __restrict__ out, int n) {
  int i = (blockIdx.x * 256 + threadIdx.x) * 4;
  if (i < n) {
#pragma unroll
    for (int q = 0; q < 4; ++q) out[i + q] = f2bf(hs[i + q]);
  }
}

// -------------------- entity gather: ea (f32), ef (bf16); reads only 2E rows
// of attention per head instead of streaming the full 192MB tensor for mean().
__global__ __launch_bounds__(256) void entity_kernel(const float* __restrict__ att,
                                                     const float* __restrict__ hs,
                                                     const int* __restrict__ head,
                                                     const int* __restrict__ tail,
                                                     float* __restrict__ ea,
                                                     bf16* __restrict__ ef) {
  int be = blockIdx.x;           // b*E + e
  int b = be / Ee;
  int hd = head[be], tl = tail[be];
  const float scl = 0.5f / (float)NHd;
  for (int l = threadIdx.x; l < Lsz; l += 256) {
    float s = 0.f;
#pragma unroll
    for (int h = 0; h < NHd; ++h) {
      const float* base = att + ((size_t)(b * NHd + h) * Lsz) * Lsz + l;
      s += base[(size_t)hd * Lsz] + base[(size_t)tl * Lsz];
    }
    ea[(size_t)be * Lsz + l] = s * scl;
  }
  for (int c = threadIdx.x; c < Hsz; c += 256) {
    float v = 0.5f * (hs[((size_t)(b * Lsz + hd)) * Hsz + c] +
                      hs[((size_t)(b * Lsz + tl)) * Hsz + c]);
    ef[(size_t)be * Hsz + c] = f2bf(v);
  }
}

// ------------------- per-pair: normalized ha (bf16), entity & distance columns
__global__ __launch_bounds__(256) void pair_kernel(const float* __restrict__ ea,
                                                   const bf16* __restrict__ ef,
                                                   const float* __restrict__ mask,
                                                   const int* __restrict__ head,
                                                   const float* __restrict__ dis,
                                                   bf16* __restrict__ ha,
                                                   bf16* __restrict__ h_in,
                                                   bf16* __restrict__ t_in) {
  int bp = blockIdx.x;           // b*Ppad + p
  int b = bp / Ppad, p = bp % Ppad;
  int t = threadIdx.x;
  bf16* harow = ha + (size_t)bp * Lsz;
  if (p >= Pp) {                 // zero pad rows: WMMA M-padding is benign
    for (int l = t; l < Lsz; l += 256) harow[l] = f2bf(0.f);
    return;
  }
  int i0 = p / (Ee - 1);
  int rr = p % (Ee - 1);
  int i1 = rr + (rr >= i0 ? 1 : 0);

  const float* e0 = ea + ((size_t)b * Ee + i0) * Lsz;
  const float* e1 = ea + ((size_t)b * Ee + i1) * Lsz;
  const float* mk = mask + (size_t)b * Lsz;
  float w[4]; float s = 0.f;
#pragma unroll
  for (int q = 0; q < 4; ++q) {
    int l = t + q * 256;
    w[q] = e0[l] * e1[l] * mk[l];
    s += w[q];
  }
  __shared__ float red[256];
  red[t] = s; __syncthreads();
  for (int off = 128; off > 0; off >>= 1) {
    if (t < off) red[t] += red[t + off];
    __syncthreads();
  }
  float inv = 1.f / (red[0] + 1e-20f);
#pragma unroll
  for (int q = 0; q < 4; ++q) harow[t + q * 256] = f2bf(w[q] * inv);

  size_t n = (size_t)b * Pp + p;
  const bf16* f0 = ef + ((size_t)b * Ee + i0) * Hsz;
  const bf16* f1 = ef + ((size_t)b * Ee + i1) * Hsz;
  for (int c = t; c < Hsz; c += 256) {
    h_in[n * INf + c] = f0[c];
    t_in[n * INf + c] = f1[c];
  }
  int x = head[b * Ee + i0] - head[b * Ee + i1];
  if (x < 0) x = -x;
  int bucket = (x < 2) ? 0 : (31 - __clz(x));
  if (bucket > 9) bucket = 9;
  if (t < Dd) {
    float dv = dis[bucket * Dd + t];
    h_in[n * INf + 2 * Hsz + t] = f2bf(dv);
    t_in[n * INf + 2 * Hsz + t] = f2bf(-dv);
  }
}

// ----- shared tile loaders (no guards: all A buffers are M-padded to 128) -----
DEV void load_a_tile_async(const bf16* A, size_t lda, int m0, int k0, bf16* As, int t) {
#pragma unroll
  for (int c = t; c < BM * BKL / 8; c += 256) {     // 4 x b128 per thread
    int row = c >> 3, col = (c & 7) * 8;
    async_copy_b128(lds_off_of(&As[row * BKL + col]),
                    A + (size_t)(m0 + row) * lda + k0 + col);
  }
}
DEV void load_b_tile_f32(const float* W, int ldb, int k0, int n0, bf16* Bs, int t) {
#pragma unroll
  for (int c = t; c < BN * BKL / 4; c += 256) {     // transpose to [n][k]
    int k = c >> 4, n4 = (c & 15) * 4;
    const float* src = W + (size_t)(k0 + k) * ldb + n0 + n4;
#pragma unroll
    for (int i = 0; i < 4; ++i) Bs[(n4 + i) * BKL + k] = f2bf(src[i]);
  }
}

// ------------------------------- GEMM: info = ha @ hs, dual write to h_in/t_in
__global__ __launch_bounds__(256) void gemm_info_kernel(const bf16* __restrict__ ha,
                                                        const bf16* __restrict__ hsb,
                                                        bf16* __restrict__ h_in,
                                                        bf16* __restrict__ t_in) {
  int b = blockIdx.z;
  int m0 = blockIdx.x * BM, n0 = blockIdx.y * BN;
  const bf16* A  = ha + (size_t)b * Ppad * Lsz;
  const bf16* Bm = hsb + (size_t)b * Lsz * Hsz;
  __shared__ __align__(32) bf16 As[BM * BKL];
  __shared__ __align__(32) bf16 Bs[BN * BKL];
  int t = threadIdx.x;
  int lane = t & 31, wid = t >> 5;
  int kh = lane >> 4, lr = lane & 15;
  int wm = wid >> 1, wn = wid & 1;
  v8f acc[2][2] = {};
  for (int k0 = 0; k0 < Lsz; k0 += BKL) {
    __syncthreads();
    load_a_tile_async(A, Lsz, m0, k0, As, t);
    for (int c = t; c < BN * BKL / 4; c += 256) {   // bf16 B, transpose
      int k = c >> 4, n4 = (c & 15) * 4;
      const bf16* src = Bm + (size_t)(k0 + k) * Hsz + n0 + n4;
#pragma unroll
      for (int i = 0; i < 4; ++i) Bs[(n4 + i) * BKL + k] = src[i];
    }
    async_wait0();
    __syncthreads();
#pragma unroll
    for (int ks = 0; ks < 2; ++ks) {
      v16bf af[2], bfr[2];
#pragma unroll
      for (int tm = 0; tm < 2; ++tm) af[tm] = load_a_frag(As, wm * 32 + tm * 16 + lr, ks, kh);
#pragma unroll
      for (int tn = 0; tn < 2; ++tn) bfr[tn] = load_b_frag(Bs, wn * 32 + tn * 16 + lr, ks, kh);
#pragma unroll
      for (int tm = 0; tm < 2; ++tm)
#pragma unroll
        for (int tn = 0; tn < 2; ++tn)
          acc[tm][tn] = __builtin_amdgcn_wmma_f32_16x16x32_bf16(
              false, af[tm], false, bfr[tn], (short)0, acc[tm][tn], false, false);
    }
  }
#pragma unroll
  for (int tm = 0; tm < 2; ++tm)
#pragma unroll
    for (int tn = 0; tn < 2; ++tn)
#pragma unroll
      for (int r = 0; r < 8; ++r) {
        int m = m0 + wm * 32 + tm * 16 + kh * 8 + r;
        int col = n0 + wn * 32 + tn * 16 + lr;
        if (m < Pp) {
          size_t n = (size_t)b * Pp + m;
          bf16 v = f2bf(acc[tm][tn][r]);
          h_in[n * INf + Hsz + col] = v;
          t_in[n * INf + Hsz + col] = v;
        }
      }
}

// --------------- fused GLU gate: uv = sigmoid(x@uW+ub) * (x@vW+vb)  (bf16 out)
__global__ __launch_bounds__(256) void gemm_gate_kernel(const bf16* __restrict__ A,
                                                        const float* __restrict__ Wu,
                                                        const float* __restrict__ Wv,
                                                        const float* __restrict__ bu,
                                                        const float* __restrict__ bv,
                                                        bf16* __restrict__ out) {
  int m0 = blockIdx.x * BM, n0 = blockIdx.y * BN;
  __shared__ __align__(32) bf16 As[BM * BKL];
  __shared__ __align__(32) bf16 Bu[BN * BKL];
  __shared__ __align__(32) bf16 Bv[BN * BKL];
  int t = threadIdx.x;
  int lane = t & 31, wid = t >> 5;
  int kh = lane >> 4, lr = lane & 15;
  int wm = wid >> 1, wn = wid & 1;
  v8f au[2][2] = {}, av[2][2] = {};
  for (int k0 = 0; k0 < INf; k0 += BKL) {
    if (k0 + BKL < INf) __builtin_prefetch(Wu + (size_t)(k0 + BKL) * K3 + n0, 0, 1);
    __syncthreads();
    load_a_tile_async(A, INf, m0, k0, As, t);
    load_b_tile_f32(Wu, K3, k0, n0, Bu, t);
    load_b_tile_f32(Wv, K3, k0, n0, Bv, t);
    async_wait0();
    __syncthreads();
#pragma unroll
    for (int ks = 0; ks < 2; ++ks) {
      v16bf af[2], bu_[2], bv_[2];
#pragma unroll
      for (int tm = 0; tm < 2; ++tm) af[tm] = load_a_frag(As, wm * 32 + tm * 16 + lr, ks, kh);
#pragma unroll
      for (int tn = 0; tn < 2; ++tn) {
        bu_[tn] = load_b_frag(Bu, wn * 32 + tn * 16 + lr, ks, kh);
        bv_[tn] = load_b_frag(Bv, wn * 32 + tn * 16 + lr, ks, kh);
      }
#pragma unroll
      for (int tm = 0; tm < 2; ++tm)
#pragma unroll
        for (int tn = 0; tn < 2; ++tn) {
          au[tm][tn] = __builtin_amdgcn_wmma_f32_16x16x32_bf16(
              false, af[tm], false, bu_[tn], (short)0, au[tm][tn], false, false);
          av[tm][tn] = __builtin_amdgcn_wmma_f32_16x16x32_bf16(
              false, af[tm], false, bv_[tn], (short)0, av[tm][tn], false, false);
        }
    }
  }
#pragma unroll
  for (int tm = 0; tm < 2; ++tm)
#pragma unroll
    for (int tn = 0; tn < 2; ++tn)
#pragma unroll
      for (int r = 0; r < 8; ++r) {
        int m = m0 + wm * 32 + tm * 16 + kh * 8 + r;
        int col = n0 + wn * 32 + tn * 16 + lr;
        if (m < Nrow) {
          float u = au[tm][tn][r] + bu[col];
          float v = av[tm][tn][r] + bv[col];
          float g = v * (1.f / (1.f + __expf(-u)));
          out[(size_t)m * K3 + col] = f2bf(g);
        }
      }
}

// ----------------------- out-proj GEMM: all = tanh(uv @ oW + ob)  (bf16 out)
__global__ __launch_bounds__(256) void gemm_tanh_kernel(const bf16* __restrict__ A,
                                                        const float* __restrict__ W,
                                                        const float* __restrict__ bias,
                                                        bf16* __restrict__ out) {
  int m0 = blockIdx.x * BM, n0 = blockIdx.y * BN;
  __shared__ __align__(32) bf16 As[BM * BKL];
  __shared__ __align__(32) bf16 Bs[BN * BKL];
  int t = threadIdx.x;
  int lane = t & 31, wid = t >> 5;
  int kh = lane >> 4, lr = lane & 15;
  int wm = wid >> 1, wn = wid & 1;
  v8f acc[2][2] = {};
  for (int k0 = 0; k0 < K3; k0 += BKL) {
    if (k0 + BKL < K3) __builtin_prefetch(W + (size_t)(k0 + BKL) * Hsz + n0, 0, 1);
    __syncthreads();
    load_a_tile_async(A, K3, m0, k0, As, t);
    load_b_tile_f32(W, Hsz, k0, n0, Bs, t);
    async_wait0();
    __syncthreads();
#pragma unroll
    for (int ks = 0; ks < 2; ++ks) {
      v16bf af[2], bfr[2];
#pragma unroll
      for (int tm = 0; tm < 2; ++tm) af[tm] = load_a_frag(As, wm * 32 + tm * 16 + lr, ks, kh);
#pragma unroll
      for (int tn = 0; tn < 2; ++tn) bfr[tn] = load_b_frag(Bs, wn * 32 + tn * 16 + lr, ks, kh);
#pragma unroll
      for (int tm = 0; tm < 2; ++tm)
#pragma unroll
        for (int tn = 0; tn < 2; ++tn)
          acc[tm][tn] = __builtin_amdgcn_wmma_f32_16x16x32_bf16(
              false, af[tm], false, bfr[tn], (short)0, acc[tm][tn], false, false);
    }
  }
#pragma unroll
  for (int tm = 0; tm < 2; ++tm)
#pragma unroll
    for (int tn = 0; tn < 2; ++tn)
#pragma unroll
      for (int r = 0; r < 8; ++r) {
        int m = m0 + wm * 32 + tm * 16 + kh * 8 + r;
        int col = n0 + wn * 32 + tn * 16 + lr;
        if (m < Nrow) out[(size_t)m * Hsz + col] = f2bf(tanhf(acc[tm][tn][r] + bias[col]));
      }
}

// ------------------------------------------------- pred init: bias broadcast
__global__ __launch_bounds__(256) void init_pred_kernel(const float* __restrict__ bias,
                                                        float* __restrict__ out) {
  int i = blockIdx.x * 256 + threadIdx.x;
  if (i < Nrow * Rr) out[i] = bias[i % Rr];
}

// ---- GroupLinear as GEMM with on-the-fly A = a_i * b_j, K split over groups
__global__ __launch_bounds__(256) void gemm_cls_kernel(const bf16* __restrict__ ah,
                                                       const bf16* __restrict__ at,
                                                       const float* __restrict__ W,
                                                       float* __restrict__ out) {
  int m0 = blockIdx.x * BM, n0 = blockIdx.y * BN, g = blockIdx.z;
  __shared__ __align__(32) bf16 As[BM * BKL];
  __shared__ __align__(32) bf16 Bs[BN * BKL];
  int t = threadIdx.x;
  int lane = t & 31, wid = t >> 5;
  int kh = lane >> 4, lr = lane & 15;
  int wm = wid >> 1, wn = wid & 1;
  v8f acc[2][2] = {};
  for (int kl = 0; kl < BSb * BSb; kl += BKL) {   // K-local = i*64 + j
    int ii = kl >> 6;
    __syncthreads();
    {   // build A tile: As[m][j] = a[n,g,ii] * b[n,g,j]   (rows padded)
      int m = t >> 1, hh = t & 1;
      int n = m0 + m;
      float a = bf2f(ah[(size_t)n * Hsz + g * BSb + ii]);
      const bf16* bp = at + (size_t)n * Hsz + g * BSb + hh * 32;
      bf16* dst = &As[m * BKL + hh * 32];
#pragma unroll
      for (int q8 = 0; q8 < 4; ++q8) {
        v8bf bv8 = *(const v8bf*)(bp + q8 * 8);
#pragma unroll
        for (int q = 0; q < 8; ++q) dst[q8 * 8 + q] = f2bf(a * bf2f(bv8[q]));
      }
    }
    for (int c = t; c < BN * BKL / 4; c += 256) {
      int k = c >> 4, n4 = (c & 15) * 4;
      const float* src = W + (size_t)(g * (BSb * BSb) + kl + k) * Rr + n0 + n4;
#pragma unroll
      for (int i = 0; i < 4; ++i)
        Bs[(n4 + i) * BKL + k] = f2bf((n0 + n4 + i < Rr) ? src[i] : 0.f);
    }
    __syncthreads();
#pragma unroll
    for (int ks = 0; ks < 2; ++ks) {
      v16bf af[2], bfr[2];
#pragma unroll
      for (int tm = 0; tm < 2; ++tm) af[tm] = load_a_frag(As, wm * 32 + tm * 16 + lr, ks, kh);
#pragma unroll
      for (int tn = 0; tn < 2; ++tn) bfr[tn] = load_b_frag(Bs, wn * 32 + tn * 16 + lr, ks, kh);
#pragma unroll
      for (int tm = 0; tm < 2; ++tm)
#pragma unroll
        for (int tn = 0; tn < 2; ++tn)
          acc[tm][tn] = __builtin_amdgcn_wmma_f32_16x16x32_bf16(
              false, af[tm], false, bfr[tn], (short)0, acc[tm][tn], false, false);
    }
  }
#pragma unroll
  for (int tm = 0; tm < 2; ++tm)
#pragma unroll
    for (int tn = 0; tn < 2; ++tn)
#pragma unroll
      for (int r = 0; r < 8; ++r) {
        int m = m0 + wm * 32 + tm * 16 + kh * 8 + r;
        int col = n0 + wn * 32 + tn * 16 + lr;
        if (m < Nrow && col < Rr) atomicAdd(&out[(size_t)m * Rr + col], acc[tm][tn][r]);
      }
}

extern "C" void kernel_launch(void* const* d_in, const int* in_sizes, int n_in,
                              void* d_out, int out_size, void* d_ws, size_t ws_size,
                              hipStream_t stream) {
  (void)in_sizes; (void)n_in; (void)out_size; (void)ws_size;
  const float* hs   = (const float*)d_in[0];
  const float* att  = (const float*)d_in[1];
  const float* mask = (const float*)d_in[2];
  const int*   head = (const int*)d_in[3];
  const int*   tail = (const int*)d_in[4];
  const float* uW_h = (const float*)d_in[5];
  const float* ub_h = (const float*)d_in[6];
  const float* vW_h = (const float*)d_in[7];
  const float* vb_h = (const float*)d_in[8];
  const float* oW_h = (const float*)d_in[9];
  const float* ob_h = (const float*)d_in[10];
  const float* uW_t = (const float*)d_in[11];
  const float* ub_t = (const float*)d_in[12];
  const float* vW_t = (const float*)d_in[13];
  const float* vb_t = (const float*)d_in[14];
  const float* oW_t = (const float*)d_in[15];
  const float* ob_t = (const float*)d_in[16];
  const float* clsW = (const float*)d_in[17];
  const float* clsb = (const float*)d_in[18];
  const float* dis  = (const float*)d_in[19];
  float* out = (float*)d_out;

  char* w = (char*)d_ws;
  auto take = [&](size_t bytes) {
    char* p = w;
    w += (bytes + 255) & ~size_t(255);
    return p;
  };
  bf16*  hs_bf = (bf16*) take((size_t)Bsz * Lsz * Hsz * 2);
  float* ea    = (float*)take((size_t)Bsz * Ee * Lsz * 4);
  bf16*  ef    = (bf16*) take((size_t)Bsz * Ee * Hsz * 2);
  bf16*  ha    = (bf16*) take((size_t)Bsz * Ppad * Lsz * 2);
  bf16*  h_in  = (bf16*) take((size_t)Nrowp * INf * 2);
  bf16*  t_in  = (bf16*) take((size_t)Nrowp * INf * 2);
  bf16*  uv_h  = (bf16*) take((size_t)Nrowp * K3 * 2);
  bf16*  uv_t  = (bf16*) take((size_t)Nrowp * K3 * 2);
  bf16*  all_h = (bf16*) take((size_t)Nrowp * Hsz * 2);
  bf16*  all_t = (bf16*) take((size_t)Nrowp * Hsz * 2);

  dim3 blk(256);
  int nhs = Bsz * Lsz * Hsz;
  prep_hs_kernel<<<(nhs / 4 + 255) / 256, blk, 0, stream>>>(hs, hs_bf, nhs);
  entity_kernel<<<Bsz * Ee, blk, 0, stream>>>(att, hs, head, tail, ea, ef);
  pair_kernel<<<Bsz * Ppad, blk, 0, stream>>>(ea, ef, mask, head, dis, ha, h_in, t_in);
  gemm_info_kernel<<<dim3(Ppad / BM, Hsz / BN, Bsz), blk, 0, stream>>>(
      ha, hs_bf, h_in, t_in);
  gemm_gate_kernel<<<dim3(Nrowp / BM, K3 / BN), blk, 0, stream>>>(
      h_in, uW_h, vW_h, ub_h, vb_h, uv_h);
  gemm_gate_kernel<<<dim3(Nrowp / BM, K3 / BN), blk, 0, stream>>>(
      t_in, uW_t, vW_t, ub_t, vb_t, uv_t);
  gemm_tanh_kernel<<<dim3(Nrowp / BM, Hsz / BN), blk, 0, stream>>>(
      uv_h, oW_h, ob_h, all_h);
  gemm_tanh_kernel<<<dim3(Nrowp / BM, Hsz / BN), blk, 0, stream>>>(
      uv_t, oW_t, ob_t, all_t);
  init_pred_kernel<<<(Nrow * Rr + 255) / 256, blk, 0, stream>>>(clsb, out);
  gemm_cls_kernel<<<dim3(Nrowp / BM, (Rr + BN - 1) / BN, Gg), blk, 0, stream>>>(
      all_h, all_t, clsW, out);
}
